// SortPool_55808805044904
// MI455X (gfx1250) — matrix operand order repaired and checked by reading, not compiled, not run on hardware
//
#include <hip/hip_runtime.h>

// ---------------- problem constants (from reference) ----------------
#define G        2048
#define NPG      64            // nodes per graph
#define HID      128
#define CLASSES  10
#define K_TOP    30
#define LAYERS   4
#define DEG      16
#define N_NODES  (G * NPG)      // 131072
#define N_EDGES  (N_NODES * DEG)// 2097152
#define EPG      (NPG * DEG)    // 1024 edges per graph (contiguous per ref)
#define KCAT     256            // fused K of [mean | x] GEMM
#define KD1      (K_TOP * HID)  // 3840, lin1 K

// dynamic LDS layout for sage_layer (bytes)
#define SM_AGG   0
#define SM_CNT   (NPG * HID * 4)             // 32768
#define SM_MEAN  (SM_CNT + NPG * 4)          // 33024
#define SM_W     (SM_MEAN + NPG * HID * 2)   // 49408
#define SM_TOTAL (SM_W + HID * KCAT * 2)     // 114944

// ---------------- CDNA5 WMMA types ----------------
typedef __attribute__((ext_vector_type(16))) __bf16 v16bf;
typedef __attribute__((ext_vector_type(8)))  float  v8f;

union V16 { v16bf v; uint4 u[2]; };

__device__ __forceinline__ unsigned short f2bf(float f) {
  unsigned int u = __float_as_uint(f);
  unsigned int r = u + 0x7FFFu + ((u >> 16) & 1u);   // RNE
  return (unsigned short)(r >> 16);
}
__device__ __forceinline__ float bf2f(unsigned short s) {
  return __uint_as_float(((unsigned int)s) << 16);
}

// async global->LDS copy of 16B, via inline asm (portable across toolchains;
// clang builtin arity differs between ROCm 7.2 and upstream). Tracked by
// ASYNCcnt; pair with s_wait_asynccnt before consuming the LDS data.
__device__ __forceinline__ void async_copy_b128(const void* gsrc, void* ldst) {
  unsigned long long ga = (unsigned long long)gsrc;
  unsigned la = (unsigned)(unsigned long long)ldst;   // flat-LDS low 32 bits
  asm volatile("global_load_async_to_lds_b128 %0, %1, off"
               :: "v"(la), "v"(ga)
               : "memory");
}
__device__ __forceinline__ void wait_asynccnt0() {
  asm volatile("s_wait_asynccnt 0" ::: "memory");
}

// ---------------- prep kernels ----------------
__global__ void cvt_f32_bf16(const float* __restrict__ in,
                             unsigned short* __restrict__ out, int n) {
  for (int i = blockIdx.x * blockDim.x + threadIdx.x; i < n;
       i += gridDim.x * blockDim.x)
    out[i] = f2bf(in[i]);
}

// Wt[layer][n][k] (bf16, n<128, k<256): k<128 -> w_l[k][n], else w_r[k-128][n]
__global__ void build_sage_wt(const float* __restrict__ w1l,
                              const float* __restrict__ w1r,
                              const float* __restrict__ wsl,
                              const float* __restrict__ wsr,
                              unsigned short* __restrict__ wt) {
  int total = LAYERS * HID * KCAT;
  for (int i = blockIdx.x * blockDim.x + threadIdx.x; i < total;
       i += gridDim.x * blockDim.x) {
    int layer = i / (HID * KCAT);
    int rem   = i % (HID * KCAT);
    int n = rem / KCAT, k = rem % KCAT;
    float v;
    if (layer == 0) {
      v = (k < HID) ? w1l[k * HID + n] : w1r[(k - HID) * HID + n];
    } else {
      const float* wl = wsl + (size_t)(layer - 1) * HID * HID;
      const float* wr = wsr + (size_t)(layer - 1) * HID * HID;
      v = (k < HID) ? wl[k * HID + n] : wr[(k - HID) * HID + n];
    }
    wt[i] = f2bf(v);
  }
}

// lin1_wt[n][k] = lin1_w[k][n]  (n<128, k<3840)
__global__ void build_lin1_wt(const float* __restrict__ w,
                              unsigned short* __restrict__ wt) {
  int total = HID * KD1;
  for (int i = blockIdx.x * blockDim.x + threadIdx.x; i < total;
       i += gridDim.x * blockDim.x) {
    int n = i / KD1, k = i % KD1;
    wt[i] = f2bf(w[(size_t)k * HID + n]);
  }
}

// ---------------- fused SAGE layer ----------------
// one workgroup per graph:
//   [async-stage 64KB weights -> LDS]  overlapped with
//   LDS f32 segment-sum -> mean(bf16) -> WMMA [mean|x] @ [wl;wr] -> ReLU
__global__ __launch_bounds__(256) void sage_layer(
    const unsigned short* __restrict__ hin,   // [N_NODES][128] bf16
    unsigned short* __restrict__ hout,        // [N_NODES][128] bf16
    const int* __restrict__ src, const int* __restrict__ dst,
    const unsigned short* __restrict__ wt,    // [128][256] bf16 (this layer)
    const float* __restrict__ bias)           // [128]
{
  extern __shared__ char smem[];
  float*          agg_s  = (float*)(smem + SM_AGG);          // 64x128 f32
  float*          cnt_s  = (float*)(smem + SM_CNT);          // 64 f32
  unsigned short* mean_s = (unsigned short*)(smem + SM_MEAN);// 64x128 bf16
  unsigned short* wlds   = (unsigned short*)(smem + SM_W);   // 128x256 bf16

  const int g = blockIdx.x, tid = threadIdx.x;
  const int gbase = g * NPG;
  const int wave = tid >> 5, lane = tid & 31;

  // ---- stage weights into LDS, overlapped with aggregation below ----
  // 64KB total: 256 threads x 16 steps x 16B, ASYNCcnt-tracked
  #pragma unroll
  for (int i = 0; i < 16; ++i)
    async_copy_b128(wt + tid * 8 + i * 2048, wlds + tid * 8 + i * 2048);

  for (int i = tid; i < NPG * HID; i += 256) agg_s[i] = 0.f;
  if (tid < NPG) cnt_s[tid] = 0.f;
  __syncthreads();

  // ---- segment sum over this graph's 1024 contiguous edges ----
  const int ebase = g * EPG;
  for (int e = wave; e < EPG; e += 8) {
    int s = src[ebase + e] - gbase;
    int d = dst[ebase + e] - gbase;
    const unsigned short* hp = hin + (size_t)(gbase + s) * HID + lane * 4;
    uint2 raw = *(const uint2*)hp;                  // 4 bf16 channels / lane
    int c = d * HID + lane * 4;
    atomicAdd(&agg_s[c + 0], bf2f((unsigned short)(raw.x & 0xFFFF)));
    atomicAdd(&agg_s[c + 1], bf2f((unsigned short)(raw.x >> 16)));
    atomicAdd(&agg_s[c + 2], bf2f((unsigned short)(raw.y & 0xFFFF)));
    atomicAdd(&agg_s[c + 3], bf2f((unsigned short)(raw.y >> 16)));
    if (lane == 0) atomicAdd(&cnt_s[d], 1.f);
  }
  __syncthreads();

  // ---- mean -> bf16 ----
  for (int i = tid; i < NPG * HID; i += 256) {
    int r = i >> 7;
    float m = agg_s[i] / fmaxf(cnt_s[r], 1.f);
    mean_s[i] = f2bf(m);
  }

  // weights must be resident in LDS before the GEMM phase
  wait_asynccnt0();
  __syncthreads();

  // ---- WMMA GEMM: out(64x128) = mean @ w_l + x @ w_r + b, ReLU ----
  const int rt = wave & 3;            // row tile (16 rows each)
  const int cg = wave >> 2;           // col group: 4 col tiles each
  const int lhalf = lane >> 4, l15 = lane & 15;
  const int arow = rt * 16 + l15;

  v8f acc[4] = {};

  #pragma unroll
  for (int part = 0; part < 2; ++part) {        // 0: mean@w_l  1: x@w_r
    #pragma unroll
    for (int kk = 0; kk < 4; ++kk) {            // K = 128 per part
      int k0 = kk * 32 + lhalf * 8;             // A: 8+8 bf16 chunks
      V16 A;
      if (part == 0) {
        const unsigned short* ap = mean_s + arow * HID;
        A.u[0] = *(const uint4*)(ap + k0);
        A.u[1] = *(const uint4*)(ap + k0 + 16);
      } else {
        const unsigned short* ap = hin + (size_t)(gbase + arow) * HID;
        A.u[0] = *(const uint4*)(ap + k0);
        A.u[1] = *(const uint4*)(ap + k0 + 16);
      }
      int kb = part * HID + kk * 32 + lhalf * 8;    // B: 8+8 bf16 chunks
      #pragma unroll
      for (int t = 0; t < 4; ++t) {
        int nc = (cg * 4 + t) * 16 + l15;
        const unsigned short* bp = wlds + nc * KCAT + kb;   // from LDS
        V16 B;
        B.u[0] = *(const uint4*)(bp);
        B.u[1] = *(const uint4*)(bp + 16);
        acc[t] = __builtin_amdgcn_wmma_f32_16x16x32_bf16(
            false, A.v, false, B.v, (short)0, acc[t], false, false);
      }
    }
  }

  #pragma unroll
  for (int t = 0; t < 4; ++t) {
    int nc = (cg * 4 + t) * 16 + l15;
    float b = bias[nc];
    #pragma unroll
    for (int r = 0; r < 8; ++r) {
      int m = rt * 16 + r + lhalf * 8;          // C/D VGPR->row mapping
      float v = fmaxf(acc[t][r] + b, 0.f);
      hout[(size_t)(gbase + m) * HID + nc] = f2bf(v);
    }
  }
}

// ---------------- sort-pool (per graph: rank by counting, gather top-30) ----
__global__ __launch_bounds__(128) void sortpool(
    const unsigned short* __restrict__ h,     // final [N_NODES][128]
    unsigned short* __restrict__ p)           // [G][3840]
{
  __shared__ float keys[NPG];
  __shared__ int   sel[K_TOP];
  const int g = blockIdx.x, tid = threadIdx.x;
  const int gbase = g * NPG;
  if (tid < NPG)
    keys[tid] = bf2f(h[(size_t)(gbase + tid) * HID + (HID - 1)]);
  __syncthreads();
  if (tid < NPG) {
    float k = keys[tid];
    int rank = 0;
    for (int j = 0; j < NPG; ++j) {
      float kj = keys[j];
      rank += (kj > k) || (kj == k && j < tid);   // stable desc (argsort(-v))
    }
    if (rank < K_TOP) sel[rank] = tid;
  }
  __syncthreads();
  for (int s = 0; s < K_TOP; ++s) {
    int node = sel[s];
    p[(size_t)g * KD1 + s * HID + tid] =
        h[(size_t)(gbase + node) * HID + tid];
  }
}

// ---------------- lin1: [2048,3840] @ [3840,128] bf16 WMMA ----------------
__global__ __launch_bounds__(256) void lin1_gemm(
    const unsigned short* __restrict__ p,     // [G][3840]
    const unsigned short* __restrict__ wt,    // [128][3840]
    const float* __restrict__ bias,           // [128]
    unsigned short* __restrict__ q)           // [G][128]
{
  const int tid = threadIdx.x;
  const int wave = tid >> 5, lane = tid & 31;
  const int rt = wave & 3, cg = wave >> 2;
  const int lhalf = lane >> 4, l15 = lane & 15;
  const int row = blockIdx.x * 64 + rt * 16 + l15;   // graph index

  if (tid == 0) __builtin_prefetch(wt);             // global_prefetch_b8

  v8f acc[4] = {};
  const unsigned short* ap = p + (size_t)row * KD1;

  for (int kk = 0; kk < KD1 / 32; ++kk) {            // 120 K-steps
    int k0 = kk * 32 + lhalf * 8;
    V16 A;
    A.u[0] = *(const uint4*)(ap + k0);
    A.u[1] = *(const uint4*)(ap + k0 + 16);
    int kb = kk * 32 + lhalf * 8;
    #pragma unroll
    for (int t = 0; t < 4; ++t) {
      int nc = (cg * 4 + t) * 16 + l15;
      const unsigned short* bp = wt + (size_t)nc * KD1 + kb;
      V16 B;
      B.u[0] = *(const uint4*)(bp);
      B.u[1] = *(const uint4*)(bp + 16);
      acc[t] = __builtin_amdgcn_wmma_f32_16x16x32_bf16(
          false, A.v, false, B.v, (short)0, acc[t], false, false);
    }
  }

  #pragma unroll
  for (int t = 0; t < 4; ++t) {
    int nc = (cg * 4 + t) * 16 + l15;
    float b = bias[nc];
    #pragma unroll
    for (int r = 0; r < 8; ++r) {
      int m = blockIdx.x * 64 + rt * 16 + r + lhalf * 8;
      q[(size_t)m * HID + nc] = f2bf(fmaxf(acc[t][r] + b, 0.f));
    }
  }
}

// ---------------- lin2 + log_softmax (tiny) ----------------
__global__ __launch_bounds__(32) void lin2_logsoftmax(
    const unsigned short* __restrict__ q,    // [G][128]
    const float* __restrict__ w,             // [128][10]
    const float* __restrict__ b,             // [10]
    float* __restrict__ out)                 // [G][10]
{
  __shared__ float logit[CLASSES];
  __shared__ float red[2];
  const int g = blockIdx.x, tid = threadIdx.x;
  if (tid < CLASSES) {
    float acc = b[tid];
    const unsigned short* qp = q + (size_t)g * HID;
    for (int k = 0; k < HID; ++k) acc += bf2f(qp[k]) * w[k * CLASSES + tid];
    logit[tid] = acc;
  }
  __syncthreads();
  if (tid == 0) {
    float mx = -1e30f;
    for (int c = 0; c < CLASSES; ++c) mx = fmaxf(mx, logit[c]);
    float s = 0.f;
    for (int c = 0; c < CLASSES; ++c) s += __expf(logit[c] - mx);
    red[0] = mx; red[1] = __logf(s);
  }
  __syncthreads();
  if (tid < CLASSES)
    out[(size_t)g * CLASSES + tid] = logit[tid] - red[0] - red[1];
}

// ---------------- launcher ----------------
extern "C" void kernel_launch(void* const* d_in, const int* in_sizes, int n_in,
                              void* d_out, int out_size, void* d_ws, size_t ws_size,
                              hipStream_t stream) {
  const float* x      = (const float*)d_in[0];
  const int*   ei     = (const int*)d_in[1];
  const int*   src    = ei;
  const int*   dst    = ei + N_EDGES;
  const float* w1l    = (const float*)d_in[3];
  const float* b1l    = (const float*)d_in[4];
  const float* w1r    = (const float*)d_in[5];
  const float* wsl    = (const float*)d_in[6];
  const float* bsl    = (const float*)d_in[7];
  const float* wsr    = (const float*)d_in[8];
  const float* lin1w  = (const float*)d_in[9];
  const float* lin1b  = (const float*)d_in[10];
  const float* lin2w  = (const float*)d_in[11];
  const float* lin2b  = (const float*)d_in[12];

  // workspace carving (bytes, 256-aligned)
  char* ws = (char*)d_ws;
  size_t off = 0;
  auto take = [&](size_t bytes) {
    char* ptr = ws + off;
    off = (off + bytes + 255) & ~(size_t)255;
    return ptr;
  };
  unsigned short* hA      = (unsigned short*)take((size_t)N_NODES * HID * 2);
  unsigned short* hB      = (unsigned short*)take((size_t)N_NODES * HID * 2);
  unsigned short* wt_sage = (unsigned short*)take((size_t)LAYERS * HID * KCAT * 2);
  unsigned short* wt_lin1 = (unsigned short*)take((size_t)HID * KD1 * 2);
  unsigned short* pbuf    = (unsigned short*)take((size_t)G * KD1 * 2);
  unsigned short* qbuf    = (unsigned short*)take((size_t)G * HID * 2);
  (void)ws_size; (void)in_sizes; (void)n_in; (void)out_size;

  cvt_f32_bf16 <<<4096, 256, 0, stream>>>(x, hA, N_NODES * HID);
  build_sage_wt<<<512, 256, 0, stream>>>(w1l, w1r, wsl, wsr, wt_sage);
  build_lin1_wt<<<1024, 256, 0, stream>>>(lin1w, wt_lin1);

  unsigned short* hin = hA;
  unsigned short* hout = hB;
  for (int layer = 0; layer < LAYERS; ++layer) {
    const float* bias = (layer == 0) ? b1l : (bsl + (size_t)(layer - 1) * HID);
    sage_layer<<<G, 256, SM_TOTAL, stream>>>(hin, hout, src, dst,
                                             wt_sage + (size_t)layer * HID * KCAT,
                                             bias);
    unsigned short* t = hin; hin = hout; hout = t;
  }
  // final activations are in `hin` after 4 swaps

  sortpool       <<<G, 128, 0, stream>>>(hin, pbuf);
  lin1_gemm      <<<G / 64, 256, 0, stream>>>(pbuf, wt_lin1, lin1b, qbuf);
  lin2_logsoftmax<<<G, 32, 0, stream>>>(qbuf, lin2w, lin2b, (float*)d_out);
}
